// SocialLSTM_68058051772553
// MI455X (gfx1250) — matrix operand order repaired and checked by reading, not compiled, not run on hardware
//
#include <hip/hip_runtime.h>

// SocialLSTM step for MI455X (gfx1250, wave32, WMMA + async LDS staging).
//
// Pipeline (all on `stream`):
//   prep_kernel:   W_hh f32 -> f16 in ws; zero 256x128 grid-sum bins in ws
//   lstm_wmma:     W_hh(f16) staged to LDS once per block via
//                  global_load_async_to_lds_b128 (ASYNCcnt), overlapped with
//                  A-fragment construction.
//                  gates = x@W_ih^T + b  (v_wmma_f32_16x16x4_f32, exact f32)
//                        + h@W_hh^T      (4x chained v_wmma_f32_16x16x32_f16,
//                                         B fragments from LDS)
//                  -> c_new stored, h_new atomically scattered into grid bins
//   social_gather: h_out[r] = bins[cell(r)]   (float4 loads/stores)

#define N_AGENTS 262144
#define HIDDEN   128
#define NGRID    16
#define W_ELEMS  (4 * HIDDEN * HIDDEN)   // 65536 f16 = 128 KB

typedef __attribute__((ext_vector_type(16))) _Float16 v16h;
typedef __attribute__((ext_vector_type(8)))  float    v8f;
typedef __attribute__((ext_vector_type(2)))  float    v2f;

__device__ __forceinline__ float sigm(float x) {
  return __builtin_amdgcn_rcpf(1.0f + __expf(-x));
}
__device__ __forceinline__ float tanh_fast(float x) {
  return 2.0f * sigm(2.0f * x) - 1.0f;
}
__device__ __forceinline__ int grid_cell(float x, float y) {
  float cx = fminf(fmaxf(x, -10.f), 10.f);
  float cy = fminf(fmaxf(y, -10.f), 10.f);
  int ix = (int)floorf((cx + 10.f) * 0.8f);   // * NGRID / 20
  int iy = (int)floorf((cy + 10.f) * 0.8f);
  ix = min(max(ix, 0), NGRID - 1);
  iy = min(max(iy, 0), NGRID - 1);
  return ix * NGRID + iy;
}

// ---- prep: convert W_hh (512x128 f32) to f16, zero grid bins --------------
__global__ __launch_bounds__(256)
void prep_kernel(const float* __restrict__ W_hh,
                 _Float16* __restrict__ w16,
                 float* __restrict__ sums) {
  const int i = blockIdx.x * 256 + threadIdx.x;   // grid = 256 blocks -> 65536
  w16[i] = (_Float16)W_hh[i];                     // 512*128 = 65536 elements
  if (i < NGRID * NGRID * HIDDEN) sums[i] = 0.0f; // 32768 bins
}

// ---- main LSTM-cell kernel ------------------------------------------------
// Block = 512 threads (16 waves), each wave owns 16 agent rows (256 rows/blk).
// W_hh(f16) is async-DMA'd into LDS once per block; every B fragment is then
// a 32B ds_load. Per t-step (16 hidden columns) each wave computes the
// i/f/g/o gate tiles:
//   1 x v_wmma_f32_16x16x4_f32   : bias + x @ W_ih^T (K=3 packed into K=4)
//   4 x v_wmma_f32_16x16x32_f16  : h @ W_hh^T (K=128), B from LDS
// then applies the cell nonlinearity directly from the accumulators.
__global__ __launch_bounds__(512)
void lstm_wmma(const float* __restrict__ coords,
               const float* __restrict__ hidden,
               const float* __restrict__ cell,
               const float* __restrict__ W_ih,
               const float* __restrict__ b_ih,
               const float* __restrict__ b_hh,
               const _Float16* __restrict__ w16,
               float* __restrict__ out_c,
               float* __restrict__ sums) {
  __shared__ __attribute__((aligned(32))) _Float16 lds_w[W_ELEMS];  // 128 KB

  const int lane  = threadIdx.x & 31;
  const int wave  = threadIdx.x >> 5;
  const int half  = lane >> 4;     // 0: lanes 0-15, 1: lanes 16-31
  const int mrow  = lane & 15;
  const int rowBase = blockIdx.x * 256 + wave * 16;

  // ---- kick off async W_hh copy: 128 KB, 512 thr x 16B x 16 rounds -------
  {
    const uint32_t ldsbase = (uint32_t)(uintptr_t)(&lds_w[0]);
    const uint64_t gbase   = (uint64_t)(uintptr_t)w16;
#pragma unroll
    for (int rnd = 0; rnd < 16; ++rnd) {
      const uint32_t off  = (uint32_t)threadIdx.x * 16u + (uint32_t)rnd * 8192u;
      const uint32_t ldst = ldsbase + off;
      const uint64_t gsrc = gbase + off;
      asm volatile("global_load_async_to_lds_b128 %0, %1, off"
                   :: "v"(ldst), "v"(gsrc) : "memory");
    }
  }

  // ---- overlap: A fragments (f16), K = 0..127 (f32 -> f16) ----------------
  // A 16x32 in contiguous k order (B matches): lane row=mrow holds K {k0..k0+15},
  // k0 = half*16 (+ kk*32 per k-step).
  v16h a[4];
  {
    const float* hrow = hidden + (size_t)(rowBase + mrow) * HIDDEN;
#pragma unroll
    for (int kk = 0; kk < 4; ++kk) {
      const int k0 = kk * 32 + half * 16;
#pragma unroll
      for (int e = 0; e < 8; ++e) a[kk][e]     = (_Float16)hrow[k0 + e];
#pragma unroll
      for (int e = 0; e < 8; ++e) a[kk][8 + e] = (_Float16)hrow[k0 + 8 + e];
    }
  }

  // A' fragment for the rank-2+bias f32 WMMA (16x4, 2 VGPRs):
  // lanes 0-15 hold K=0,1 = (x, y) of row mrow; lanes 16-31 hold K=2,3 = (1,0).
  v2f a4;
  {
    const int r = rowBase + mrow;
    const float ax = coords[2 * r + 0];
    const float ay = coords[2 * r + 1];
    a4[0] = half ? 1.0f : ax;
    a4[1] = half ? 0.0f : ay;
  }

  // Grid cell for the C/D-layout rows owned by this lane (rowBase+half*8+v).
  int seg[8];
#pragma unroll
  for (int v = 0; v < 8; ++v) {
    const int r = rowBase + half * 8 + v;
    seg[v] = grid_cell(coords[2 * r + 0], coords[2 * r + 1]);
  }

  // ---- wait for the DMA, make LDS visible to all waves --------------------
  asm volatile("s_wait_asynccnt 0x0" ::: "memory");
  __syncthreads();

  const v8f zacc = {};

  for (int t = 0; t < 8; ++t) {
    v8f acc[4];   // i, f, g, o gate tiles for hidden cols t*16..t*16+15
#pragma unroll
    for (int gsel = 0; gsel < 4; ++gsel) {
      const int g = gsel * 128 + t * 16 + mrow;   // this lane's gate column
      // B' fragment (4x16 f32, 2 VGPRs): lanes 0-15 K=0,1 = W_ih[g][0..1];
      // lanes 16-31 K=2,3 = (b_ih[g]+b_hh[g], 0).
      const float wi0 = W_ih[2 * g + 0];
      const float wi1 = W_ih[2 * g + 1];
      const float bb  = b_ih[g] + b_hh[g];
      v2f b4;
      b4[0] = half ? bb   : wi0;
      b4[1] = half ? 0.0f : wi1;

      // acc = bias + x @ W_ih^T   (exact f32 WMMA, K=4 with padded zeros)
      v8f c = __builtin_amdgcn_wmma_f32_16x16x4_f32(
                  false, a4, false, b4, (short)0, zacc, false, false);

      // acc += h @ W_hh^T : lane holds B column g, 16 consecutive K values ->
      // 32B ds_load from LDS-resident row-major W_hh[g][*].
      const _Float16* wrow = &lds_w[(size_t)g * HIDDEN];
#pragma unroll
      for (int kk = 0; kk < 4; ++kk) {
        v16h b = *(const v16h*)(wrow + kk * 32 + half * 16);
        c = __builtin_amdgcn_wmma_f32_16x16x32_f16(
                false, a[kk], false, b, (short)0, c, false, false);
      }
      acc[gsel] = c;
    }

    const int j = t * 16 + mrow;   // hidden column for this lane
#pragma unroll
    for (int v = 0; v < 8; ++v) {
      const int r = rowBase + half * 8 + v;
      const float c_old = cell[(size_t)r * HIDDEN + j];
      const float cn = sigm(acc[1][v]) * c_old
                     + sigm(acc[0][v]) * tanh_fast(acc[2][v]);
      const float hn = sigm(acc[3][v]) * tanh_fast(cn);
      out_c[(size_t)r * HIDDEN + j] = cn;
      __hip_atomic_fetch_add(&sums[seg[v] * HIDDEN + j], hn,
                             __ATOMIC_RELAXED, __HIP_MEMORY_SCOPE_AGENT);
    }
  }
}

// ---- social gather: h_out[r] = sums[cell(r)] ------------------------------
// 256 threads = 8 rows per block, 32 float4 per row.
__global__ __launch_bounds__(256)
void social_gather(const float* __restrict__ coords,
                   const float* __restrict__ sums,
                   float* __restrict__ out_h) {
  const int  tid = threadIdx.x;
  const long row = (long)blockIdx.x * 8 + (tid >> 5);
  const int  c4  = tid & 31;
  const int  s   = grid_cell(coords[2 * row], coords[2 * row + 1]);
  const float4 val = ((const float4*)(sums + (size_t)s * HIDDEN))[c4];
  ((float4*)(out_h + (size_t)row * HIDDEN))[c4] = val;
}

extern "C" void kernel_launch(void* const* d_in, const int* in_sizes, int n_in,
                              void* d_out, int out_size, void* d_ws, size_t ws_size,
                              hipStream_t stream) {
  (void)in_sizes; (void)n_in; (void)out_size; (void)ws_size;
  const float* coords = (const float*)d_in[0];
  const float* hidden = (const float*)d_in[1];
  const float* cell   = (const float*)d_in[2];
  const float* W_ih   = (const float*)d_in[3];
  const float* W_hh   = (const float*)d_in[4];
  const float* b_ih   = (const float*)d_in[5];
  const float* b_hh   = (const float*)d_in[6];

  float* out_h = (float*)d_out;                              // N*128 (social h)
  float* out_c = (float*)d_out + (size_t)N_AGENTS * HIDDEN;  // N*128 (c_new)

  _Float16* w16 = (_Float16*)d_ws;                     // 512*128 f16 = 128 KB
  float* sums = (float*)((char*)d_ws + W_ELEMS * sizeof(_Float16));
                                                       // 256*128 f32 = 128 KB

  prep_kernel<<<256, 256, 0, stream>>>(W_hh, w16, sums);
  lstm_wmma<<<N_AGENTS / 256, 512, 0, stream>>>(coords, hidden, cell,
                                                W_ih, b_ih, b_hh,
                                                w16, out_c, sums);
  social_gather<<<N_AGENTS / 8, 256, 0, stream>>>(coords, sums, out_h);
}
